// DeformableTransformer_86517821211854
// MI455X (gfx1250) — matrix operand order repaired and checked by reading, not compiled, not run on hardware
//
#include <hip/hip_runtime.h>
#include <hip/hip_bf16.h>

// Problem constants (fixed by the reference)
#define NB   4
#define LQ   8192
#define TT   16384
#define CC   256
#define MH   8
#define PP   4
#define DD   32
#define KK   256   // GEMM reduction depth == C

typedef __attribute__((ext_vector_type(16))) _Float16 v16h;
typedef __attribute__((ext_vector_type(8)))  float    v8f;

// ---------------------------------------------------------------------------
// Prep: transpose + fp32->f16 convert the small weight matrices so that the
// WMMA B-fragment load is a contiguous 16-halfword (32B) chunk per lane.
//   Wt[n][k] = W[k][n]
// Also fuse W_off|W_attn into one 64x256 matrix and b_off|b_attn into one bias.
// ---------------------------------------------------------------------------
__global__ __launch_bounds__(256)
void prep_weights(const float* __restrict__ W_val, const float* __restrict__ W_out,
                  const float* __restrict__ W_off, const float* __restrict__ W_attn,
                  const float* __restrict__ b_off, const float* __restrict__ b_attn,
                  _Float16* __restrict__ Wt_val, _Float16* __restrict__ Wt_out,
                  _Float16* __restrict__ Wt_oa, float* __restrict__ bias_oa) {
  int i = blockIdx.x * blockDim.x + threadIdx.x;
  if (i < 65536) {                       // Wt_val: 256x256
    int n = i >> 8, k = i & 255;
    Wt_val[i] = (_Float16)W_val[k * CC + n];
  } else if (i < 131072) {               // Wt_out: 256x256
    int j = i - 65536;
    int n = j >> 8, k = j & 255;
    Wt_out[j] = (_Float16)W_out[k * CC + n];
  } else if (i < 147456) {               // Wt_oa: 64x256  (rows 0..31=W_off cols, 32..63=W_attn cols)
    int j = i - 131072;
    int n = j >> 8, k = j & 255;
    float w = (n < 32) ? W_off[k * 32 + n] : W_attn[k * 32 + (n - 32)];
    Wt_oa[j] = (_Float16)w;
  } else if (i < 147520) {               // bias_oa: 64
    int j = i - 147456;
    bias_oa[j] = (j < 32) ? b_off[j] : b_attn[j - 32];
  }
}

// ---------------------------------------------------------------------------
// WMMA GEMM:  Out[Mrows x Ncols] = A[Mrows x 256](f32) * Bt^T + bias, where
// Bt is (Ncols x 256) f16 row-major over K.  Block = 256 threads = 8 waves;
// each wave computes a 16-row strip across NT column tiles (16 x NT*16).
//   grid.x = Mrows/128, grid.y = Ncols/(NT*16)
// NT=16 covers all 256 output columns in one pass -> A is streamed from HBM
// exactly once (pipeline is HBM-bound at 23.3 TB/s; redundant A reads were
// the dominant cost).
//
// B-tile addresses within one kb step share a single base pointer; the
// per-tile displacement nt*16*KK*2 bytes folds into the signed 24-bit
// instruction offset.  The tile loop is unrolled by 2 with two explicit B
// fragment buffers (ping-pong), so the next tile's loads issue while the
// current WMMA executes (staggered s_wait_loadcnt instead of full drains).
//
// Fragment layouts used here (A and B share one contiguous K order):
//  A (16-bit, 16x32): lane row = lane%16, half = lane/16,
//     elems 0..7  -> K = kb + half*16 + e
//     elems 8..15 -> K = kb + half*16 + e
//  B (16-bit, 32x16): lane col = lane%16,
//     elems 0..15 -> K = kb + half*16 + e   (one contiguous 32B chunk)
//  D: lane holds D[m][n], n = lane%16, m = (lane/16)*8 + v.
// ---------------------------------------------------------------------------
template <int NT>
__global__ __launch_bounds__(256)
void gemm_wmma_f16(const float* __restrict__ A, const _Float16* __restrict__ Bt,
                   const float* __restrict__ bias, float* __restrict__ Out,
                   int Ncols) {
  static_assert(NT % 2 == 0, "NT must be even for B double-buffering");
  const int lane = threadIdx.x & 31;
  const int wave = threadIdx.x >> 5;
  const int half = lane >> 4;
  const int l16  = lane & 15;
  const int rowBase = blockIdx.x * 128 + wave * 16;
  const int colBase = blockIdx.y * (NT * 16);

  // Per-lane base pointers; loop/tile displacements are immediates.
  const float* __restrict__ Abase =
      A + (size_t)(rowBase + l16) * KK + half * 16;
  const _Float16* __restrict__ Bbase =
      Bt + (size_t)(colBase + l16) * KK + half * 16;

  v8f acc[NT] = {};

  for (int kb = 0; kb < KK; kb += 32) {
    // ---- A fragment (fp32 load, convert to f16) ----
    const float* __restrict__ ap = Abase + kb;
    __builtin_prefetch(ap + 32, 0, 1);  // global_prefetch (speculative)
    v16h a;
#pragma unroll
    for (int e = 0; e < 8; ++e) {
      a[e]     = (_Float16)ap[e];        // K = kb + half*16 + e
      a[e + 8] = (_Float16)ap[8 + e];    // K = kb + half*16 + 8 + e
    }

    // ---- B fragments: single address register + immediate tile offsets,
    //      two-buffer ping-pong so loads overlap WMMA execution ----
    const _Float16* __restrict__ bp = Bbase + kb;
    v16h b0, b1;
#pragma unroll
    for (int e = 0; e < 16; ++e) b0[e] = bp[e];              // tile 0
#pragma unroll
    for (int nt = 0; nt < NT; nt += 2) {
#pragma unroll
      for (int e = 0; e < 16; ++e) b1[e] = bp[(nt + 1) * 16 * KK + e];
      acc[nt] = __builtin_amdgcn_wmma_f32_16x16x32_f16(
          false, a, false, b0, (short)0, acc[nt], false, false);
      if (nt + 2 < NT) {
#pragma unroll
        for (int e = 0; e < 16; ++e) b0[e] = bp[(nt + 2) * 16 * KK + e];
      }
      acc[nt + 1] = __builtin_amdgcn_wmma_f32_16x16x32_f16(
          false, a, false, b1, (short)0, acc[nt + 1], false, false);
    }
  }

  // ---- epilogue: bias add + fp32 store (row stride folds into offsets) ----
  float* __restrict__ obase =
      Out + (size_t)(rowBase + half * 8) * Ncols + colBase + l16;
#pragma unroll
  for (int nt = 0; nt < NT; ++nt) {
    const float bv = bias[colBase + nt * 16 + l16];
#pragma unroll
    for (int v = 0; v < 8; ++v) {
      obase[(size_t)v * Ncols + nt * 16] = acc[nt][v] + bv;
    }
  }
}

// ---------------------------------------------------------------------------
// Sampling: one block per (n,q).  Threads 0..31 compute per-(m,p) sampling
// position + softmax weight (P=4 softmax, redundantly per-thread), staged in
// LDS; then all 256 threads (m=tid/32, d=tid%32) do the bilinear gather over
// the value tensor (67 MB -> fully L2-resident on the 192 MB L2) and the
// weighted reduce over P, writing head_out[n,q,:].  Gathers are coalesced
// along D (32 consecutive floats = 128B per (m,p)).
// ---------------------------------------------------------------------------
__global__ __launch_bounds__(256)
void sample_deform(const float* __restrict__ offattn, const float* __restrict__ refp,
                   const float* __restrict__ value, float* __restrict__ head_out) {
  __shared__ float s_row[64];
  __shared__ int   s_i0[32];
  __shared__ float s_f[32];
  __shared__ float s_w[32];

  const int b   = blockIdx.x;        // n*Lq + q
  const int n   = b >> 13;           // / LQ
  const int tid = threadIdx.x;

  const float* __restrict__ rowp = offattn + (size_t)b * 64;
  if (tid < 64) s_row[tid] = rowp[tid];
  __syncthreads();

  if (tid < 32) {
    const int m = tid >> 2;
    // sampling position: x = ref*T + off - 0.5
    const float off = s_row[tid];
    const float x   = refp[b] * (float)TT + off - 0.5f;
    const float fl  = floorf(x);
    s_i0[tid] = (int)fl;
    s_f[tid]  = x - fl;
    // softmax over the 4 points of this head
    const float l0 = s_row[32 + m * 4 + 0];
    const float l1 = s_row[32 + m * 4 + 1];
    const float l2 = s_row[32 + m * 4 + 2];
    const float l3 = s_row[32 + m * 4 + 3];
    const float mx = fmaxf(fmaxf(l0, l1), fmaxf(l2, l3));
    const float e0 = __expf(l0 - mx), e1 = __expf(l1 - mx);
    const float e2 = __expf(l2 - mx), e3 = __expf(l3 - mx);
    s_w[tid] = __expf(s_row[32 + tid] - mx) / (e0 + e1 + e2 + e3);
  }
  __syncthreads();

  const int m = tid >> 5;
  const int d = tid & 31;
  const float* __restrict__ vb = value + (size_t)n * TT * CC + m * DD + d;

  float acc = 0.0f;
#pragma unroll
  for (int p = 0; p < PP; ++p) {
    const int   t  = m * 4 + p;
    const int   i0 = s_i0[t];
    const float f  = s_f[t];
    const float w  = s_w[t];
    const int   i1 = i0 + 1;
    const float v0 = (i0 >= 0 && i0 < TT) ? vb[(size_t)i0 * CC] : 0.0f;
    const float v1 = (i1 >= 0 && i1 < TT) ? vb[(size_t)i1 * CC] : 0.0f;
    acc += w * ((1.0f - f) * v0 + f * v1);
  }
  head_out[(size_t)b * CC + tid] = acc;
}

// ---------------------------------------------------------------------------
// Launch: prep -> value GEMM -> off/attn GEMM -> sampling -> output GEMM.
// Workspace layout (bytes):
//   [0,           67108864)  value     (N*T x 256 f32)
//   [67108864,    75497472)  offattn   (N*Lq x 64 f32)
//   [75497472,   109051904)  head_out  (N*Lq x 256 f32)
//   [109051904, +131072)     Wt_val f16
//   [.. +131072]             Wt_out f16
//   [.. +32768]              Wt_oa  f16
//   [.. +256]                bias_oa f32          total ~104.3 MiB
// ---------------------------------------------------------------------------
extern "C" void kernel_launch(void* const* d_in, const int* in_sizes, int n_in,
                              void* d_out, int out_size, void* d_ws, size_t ws_size,
                              hipStream_t stream) {
  const float* query  = (const float*)d_in[0];
  const float* refp   = (const float*)d_in[1];
  const float* input  = (const float*)d_in[2];
  // d_in[3]: input_temporal_lens (T hardcoded = 16384)
  const float* W_off  = (const float*)d_in[4];
  const float* b_off  = (const float*)d_in[5];
  const float* W_attn = (const float*)d_in[6];
  const float* b_attn = (const float*)d_in[7];
  const float* W_val  = (const float*)d_in[8];
  const float* b_val  = (const float*)d_in[9];
  const float* W_out  = (const float*)d_in[10];
  const float* b_out  = (const float*)d_in[11];

  char* ws = (char*)d_ws;
  float*    value   = (float*)(ws + 0);
  float*    offattn = (float*)(ws + 67108864ull);
  float*    head    = (float*)(ws + 75497472ull);
  _Float16* Wt_val  = (_Float16*)(ws + 109051904ull);
  _Float16* Wt_out  = (_Float16*)(ws + 109051904ull + 131072ull);
  _Float16* Wt_oa   = (_Float16*)(ws + 109051904ull + 262144ull);
  float*    bias_oa = (float*)   (ws + 109051904ull + 294912ull);

  // 1) weight transpose/convert
  prep_weights<<<(147520 + 255) / 256, 256, 0, stream>>>(
      W_val, W_out, W_off, W_attn, b_off, b_attn, Wt_val, Wt_out, Wt_oa, bias_oa);

  // 2) value projection: (N*T=65536) x 256 x 256, A streamed once (grid.y=1)
  gemm_wmma_f16<16><<<dim3(65536 / 128, 1), 256, 0, stream>>>(
      input, Wt_val, b_val, value, 256);

  // 3) fused offsets+attn logits: (N*Lq=32768) x 256 x 64
  gemm_wmma_f16<4><<<dim3(32768 / 128, 1), 256, 0, stream>>>(
      query, Wt_oa, bias_oa, offattn, 64);

  // 4) softmax + bilinear temporal sampling + weighted reduce over P
  sample_deform<<<NB * LQ, 256, 0, stream>>>(offattn, refp, value, head);

  // 5) output projection: (N*Lq=32768) x 256 x 256 -> d_out
  gemm_wmma_f16<16><<<dim3(32768 / 128, 1), 256, 0, stream>>>(
      head, Wt_out, b_out, (float*)d_out, 256);
}
